// FocusedLinearAttention_73409581023447
// MI455X (gfx1250) — matrix operand order repaired and checked by reading, not compiled, not run on hardware
//
#include <hip/hip_runtime.h>

// ---------------------------------------------------------------------------
// FocusedLinearAttention for MI455X (gfx1250, wave32, WMMA).
// B=4, N=8192, D=1024, H=16, C=64.
// Heavy contractions: v_wmma_f32_16x16x32_bf16.
// GEMM tile staging: global_load_async_to_lds_b128 (ASYNCcnt), double-buffered.
// ---------------------------------------------------------------------------

typedef __attribute__((ext_vector_type(16))) __bf16 bf16x16;
typedef __attribute__((ext_vector_type(8)))  float  f32x8;
typedef __attribute__((ext_vector_type(4)))  unsigned int u32x4;

static constexpr int Bsz = 4;
static constexpr int Nseq = 8192;
static constexpr int Dm  = 1024;
static constexpr int Hh  = 16;
static constexpr int Cc  = 64;

__device__ inline f32x8 zero8() { f32x8 z; for (int i = 0; i < 8; ++i) z[i] = 0.0f; return z; }

__device__ inline f32x8 wmma_bf16(bf16x16 a, bf16x16 b, f32x8 c) {
  return __builtin_amdgcn_wmma_f32_16x16x32_bf16(
      /*neg_a=*/false, a, /*neg_b=*/false, b,
      /*c_mod=*/(short)0, c, /*reuse_a=*/false, /*reuse_b=*/false);
}

// A-matrix fragment (16x32 bf16). LDS tile is [row][k] with `pitch` bf16 per row.
// lane<16 : K0..7 at +0, K16..23 at +32 ; lane>=16 : K8..15 at +16, K24..31 at +48.
__device__ inline bf16x16 frag_ld_a(const __bf16* lds, int row0, int pitch, int kk, int lane) {
  int r = row0 + (lane & 15);
  int half = lane >> 4;
  const char* p = (const char*)(lds + r * pitch + kk);
  union { u32x4 u[2]; bf16x16 v; } cv;
  cv.u[0] = *(const u32x4*)(p + half * 16);
  cv.u[1] = *(const u32x4*)(p + 32 + half * 16);
  return cv.v;
}

// B-matrix fragment (32x16 bf16), LDS tile stored transposed: [n][k], `pitch` bf16/row.
// lane<16: column n=lane, K0..7 at +0, K16..23 at +32; lane>=16: column n=lane-16, K8..15 at +16, K24..31 at +48.
__device__ inline bf16x16 frag_ld_b(const __bf16* lds, int col0, int pitch, int kk, int lane) {
  int r = col0 + (lane & 15);
  int half = lane >> 4;
  const char* p = (const char*)(lds + r * pitch + kk);
  union { u32x4 u[2]; bf16x16 v; } cv;
  cv.u[0] = *(const u32x4*)(p + half * 16);
  cv.u[1] = *(const u32x4*)(p + 32 + half * 16);
  return cv.v;
}

// Async global->LDS copy of 16 bytes (CDNA5 Tensor/async path, ASYNCcnt-tracked).
// LDS address = low 32 bits of the generic pointer (aperture truncation).
__device__ inline void async_cp_b128(void* lds_dst, const void* gsrc) {
  unsigned int la = (unsigned int)(unsigned long long)lds_dst;
  asm volatile("global_load_async_to_lds_b128 %0, %1, off"
               :: "v"(la), "v"(gsrc) : "memory");
}
__device__ inline void async_wait0() {
  asm volatile("s_wait_asynccnt 0x0" ::: "memory");
}

// ---------------------------------------------------------------------------
// f32 -> bf16 convert (vectorized x4)
// ---------------------------------------------------------------------------
__global__ __launch_bounds__(256) void cvt_f32_bf16_k(const float* __restrict__ src,
                                                      __bf16* __restrict__ dst, long n) {
  long i = ((long)blockIdx.x * 256 + threadIdx.x) * 4;
  if (i + 3 < n) {
    float4 v = *(const float4*)(src + i);
    union { __bf16 e[4]; uint2 u; } o;
    o.e[0] = (__bf16)v.x; o.e[1] = (__bf16)v.y; o.e[2] = (__bf16)v.z; o.e[3] = (__bf16)v.w;
    *(uint2*)(dst + i) = o.u;
  }
}

// ---------------------------------------------------------------------------
// C[M,Nn] = A[M,K](bf16,row-major) x W[Nn,K](bf16,row-major, contracted over k)
// i.e. out = A @ W^T (torch Linear). 128x128x32 tiles, 8 waves of 2x4 WMMA tiles.
// Double-buffered LDS, async global->LDS staging.
// ---------------------------------------------------------------------------
#define GBM 128
#define GBN 128
#define GBK 32
#define GP  40   // LDS row pitch in bf16 (conflict-free for b128 fragment loads)

__global__ __launch_bounds__(256) void gemm_bf16_wmma_k(const __bf16* __restrict__ A,
                                                        const __bf16* __restrict__ W,
                                                        float* __restrict__ C,
                                                        int M, int Nn, int K) {
  __shared__ __bf16 lA[2][GBM * GP];
  __shared__ __bf16 lB[2][GBN * GP];
  const int tid = threadIdx.x;
  const int lane = tid & 31, w = tid >> 5;
  const int m0 = blockIdx.y * GBM, n0 = blockIdx.x * GBN;
  const int wm = w & 3;        // 4 row groups of 32
  const int wn = w >> 2;       // 2 col groups of 64

  f32x8 acc[2][4];
  for (int i = 0; i < 2; ++i)
    for (int j = 0; j < 4; ++j) acc[i][j] = zero8();

  const int lrow = tid >> 2;            // 0..63
  const int lcol = (tid & 3) * 8;       // bf16 element offset (16B chunks)
  const __bf16* ag = A + (size_t)(m0 + lrow) * K + lcol;
  const __bf16* bg = W + (size_t)(n0 + lrow) * K + lcol;
  const int KT = K / GBK;

  // Stage one 128x32 A tile + 128x32 B tile into LDS buffer `buf` (async).
  auto stage = [&](int buf, int kt) {
    const __bf16* an = ag + (size_t)kt * GBK;
    const __bf16* bn = bg + (size_t)kt * GBK;
    async_cp_b128(&lA[buf][lrow * GP + lcol],        an);
    async_cp_b128(&lA[buf][(lrow + 64) * GP + lcol], an + (size_t)64 * K);
    async_cp_b128(&lB[buf][lrow * GP + lcol],        bn);
    async_cp_b128(&lB[buf][(lrow + 64) * GP + lcol], bn + (size_t)64 * K);
  };

  stage(0, 0);
  async_wait0();
  __syncthreads();

  for (int kt = 0; kt < KT; ++kt) {
    const int cur = kt & 1;
    if (kt + 1 < KT) {
      stage(cur ^ 1, kt + 1);
      if (kt + 2 < KT) {  // gfx1250 global_prefetch_b8 hint
        __builtin_prefetch(ag + (size_t)(kt + 2) * GBK, 0, 1);
        __builtin_prefetch(bg + (size_t)(kt + 2) * GBK, 0, 1);
      }
    }
    // Load all fragments first, then one WMMA burst (single dscnt wait).
    bf16x16 af0 = frag_ld_a(lA[cur], wm * 32,      GP, 0, lane);
    bf16x16 af1 = frag_ld_a(lA[cur], wm * 32 + 16, GP, 0, lane);
    bf16x16 bf0 = frag_ld_b(lB[cur], wn * 64,      GP, 0, lane);
    bf16x16 bf1 = frag_ld_b(lB[cur], wn * 64 + 16, GP, 0, lane);
    bf16x16 bf2 = frag_ld_b(lB[cur], wn * 64 + 32, GP, 0, lane);
    bf16x16 bf3 = frag_ld_b(lB[cur], wn * 64 + 48, GP, 0, lane);
    acc[0][0] = wmma_bf16(af0, bf0, acc[0][0]);
    acc[1][0] = wmma_bf16(af1, bf0, acc[1][0]);
    acc[0][1] = wmma_bf16(af0, bf1, acc[0][1]);
    acc[1][1] = wmma_bf16(af1, bf1, acc[1][1]);
    acc[0][2] = wmma_bf16(af0, bf2, acc[0][2]);
    acc[1][2] = wmma_bf16(af1, bf2, acc[1][2]);
    acc[0][3] = wmma_bf16(af0, bf3, acc[0][3]);
    acc[1][3] = wmma_bf16(af1, bf3, acc[1][3]);
    // Next buffer must be fully staged before anyone reads it next iteration;
    // current buffer must be fully read before it is overwritten after that.
    async_wait0();
    __syncthreads();
  }

  const int nbase = n0 + wn * 64 + (lane & 15);
  const int mhalf = (lane >> 4) * 8;
  for (int i = 0; i < 2; ++i)
    for (int j = 0; j < 4; ++j) {
      int mb = m0 + wm * 32 + i * 16 + mhalf;
      float* cp = C + (size_t)mb * Nn + nbase + j * 16;
      for (int e = 0; e < 8; ++e) cp[(size_t)e * Nn] = acc[i][j][e];
    }
}

// ---------------------------------------------------------------------------
// Focused feature map per row of 1024: f=|gelu(x)|+eps ; out = softmax(ff*log f)*mean(f)
// ---------------------------------------------------------------------------
__device__ inline float blk_reduce(float v, int op /*0=sum 1=max*/) {
  __shared__ float red[8];
  for (int o = 16; o > 0; o >>= 1) {
    float u = __shfl_xor(v, o, 32);
    v = op ? fmaxf(v, u) : (v + u);
  }
  int tid = threadIdx.x;
  __syncthreads();
  if ((tid & 31) == 0) red[tid >> 5] = v;
  __syncthreads();
  float r = red[0];
  if (op) { for (int i = 1; i < 8; ++i) r = fmaxf(r, red[i]); }
  else    { for (int i = 1; i < 8; ++i) r += red[i]; }
  return r;
}

__global__ __launch_bounds__(256) void feature_map_k(const float* __restrict__ X,
                                                     __bf16* __restrict__ Y,
                                                     const float* __restrict__ ffp) {
  const int row = blockIdx.x;
  const int tid = threadIdx.x;
  const float ff = *ffp;
  const float* x = X + (size_t)row * Dm;
  __bf16* y = Y + (size_t)row * Dm;

  float f[4], t[4];
  float lsum = 0.0f;
  for (int i = 0; i < 4; ++i) {
    float v = x[tid + 256 * i];
    float g = 0.5f * v * (1.0f + erff(v * 0.70710678118654752f));  // exact GELU
    float fv = fabsf(g) + 1e-6f;
    f[i] = fv; lsum += fv;
  }
  float mean = blk_reduce(lsum, 0) * (1.0f / 1024.0f);

  float lmax = -3.0e38f;
  for (int i = 0; i < 4; ++i) { t[i] = ff * __logf(f[i]); lmax = fmaxf(lmax, t[i]); }
  float tmax = blk_reduce(lmax, 1);

  float esum = 0.0f;
  for (int i = 0; i < 4; ++i) { t[i] = __expf(t[i] - tmax); esum += t[i]; }
  esum = blk_reduce(esum, 0);

  float scale = mean / esum;
  for (int i = 0; i < 4; ++i) y[tid + 256 * i] = (__bf16)(t[i] * scale);
}

// ---------------------------------------------------------------------------
// k_mean[bh][c] = mean_n kf[b][n][h*C+c]
// ---------------------------------------------------------------------------
__global__ __launch_bounds__(256) void kmean_k(const __bf16* __restrict__ Kf,
                                               float* __restrict__ KM) {
  const int bh = blockIdx.x;
  const int b = bh >> 4, h = bh & 15;
  const int tid = threadIdx.x;
  const int c = tid & 63, seg = tid >> 6;  // 4 segments of 2048 rows
  const __bf16* base = Kf + (size_t)b * Nseq * Dm + h * Cc + c;
  float s = 0.0f;
  for (int n = seg * 2048; n < seg * 2048 + 2048; ++n)
    s += (float)base[(size_t)n * Dm];
  __shared__ float red[256];
  red[tid] = s;
  __syncthreads();
  if (seg == 0) {
    float r = red[c] + red[c + 64] + red[c + 128] + red[c + 192];
    KM[bh * Cc + c] = r * (1.0f / (float)Nseq);
  }
}

// ---------------------------------------------------------------------------
// z[bh][n] = qf[b][n][h,:] . k_mean[bh] + eps
// ---------------------------------------------------------------------------
__global__ __launch_bounds__(256) void zinv_k(const __bf16* __restrict__ Qf,
                                              const float* __restrict__ KM,
                                              float* __restrict__ Z) {
  const int bh = blockIdx.y;
  const int b = bh >> 4, h = bh & 15;
  const int n = blockIdx.x * 256 + threadIdx.x;
  __shared__ float km[64];
  if (threadIdx.x < 64) km[threadIdx.x] = KM[bh * Cc + threadIdx.x];
  __syncthreads();
  const __bf16* q = Qf + ((size_t)(b * Nseq + n)) * Dm + h * Cc;
  float s = 1e-6f;
  for (int c = 0; c < 64; ++c) s += (float)q[c] * km[c];
  Z[(size_t)bh * Nseq + n] = s;
}

// ---------------------------------------------------------------------------
// kv[bh][c][d] = (1/N) * sum_n kf[b][n][h,c] * vp[b][n][h,d]   (64x64, WMMA over n)
// One block per head; 8 waves each own a 16(c) x 32(d) region.
// ---------------------------------------------------------------------------
__global__ __launch_bounds__(256) void kv_wmma_k(const __bf16* __restrict__ Kf,
                                                 const __bf16* __restrict__ Vp,
                                                 float* __restrict__ KV) {
  const int bh = blockIdx.x;
  const int b = bh >> 4, h = bh & 15;
  const int tid = threadIdx.x, lane = tid & 31, w = tid >> 5;
  __shared__ __bf16 lK[64 * GP];  // [c][n] pitch 40
  __shared__ __bf16 lV[64 * GP];  // [d][n]
  const int tr = w >> 1;          // c-band 0..3
  const int cb = (w & 1) * 32;    // d-base 0 or 32
  f32x8 acc0 = zero8(), acc1 = zero8();

  const int nn = tid >> 3;          // 0..31 (n within chunk)
  const int c8 = (tid & 7) * 8;     // 8 contiguous channels
  const __bf16* kg = Kf + (size_t)b * Nseq * Dm + h * Cc + c8;
  const __bf16* vg = Vp + (size_t)b * Nseq * Dm + h * Cc + c8;

  for (int n0 = 0; n0 < Nseq; n0 += 32) {
    __syncthreads();  // protect previous iteration's fragment reads
    union { u32x4 u; __bf16 e[8]; } kvv, vvv;
    kvv.u = *(const u32x4*)(kg + (size_t)(n0 + nn) * Dm);
    vvv.u = *(const u32x4*)(vg + (size_t)(n0 + nn) * Dm);
    for (int i = 0; i < 8; ++i) {
      lK[(c8 + i) * GP + nn] = kvv.e[i];   // transpose into [c][n]
      lV[(c8 + i) * GP + nn] = vvv.e[i];   // transpose into [d][n]
    }
    __syncthreads();
    bf16x16 a  = frag_ld_a(lK, tr * 16, GP, 0, lane);
    bf16x16 b0 = frag_ld_b(lV, cb,      GP, 0, lane);
    bf16x16 b1 = frag_ld_b(lV, cb + 16, GP, 0, lane);
    acc0 = wmma_bf16(a, b0, acc0);
    acc1 = wmma_bf16(a, b1, acc1);
  }

  const float inv = 1.0f / (float)Nseq;
  const int mrow = tr * 16 + (lane >> 4) * 8;
  const int col = lane & 15;
  float* out = KV + (size_t)bh * (Cc * Cc);
  for (int e = 0; e < 8; ++e) {
    out[(mrow + e) * Cc + cb + col]      = acc0[e] * inv;
    out[(mrow + e) * Cc + cb + 16 + col] = acc1[e] * inv;
  }
}

// ---------------------------------------------------------------------------
// y[b][n][h,d] = (qh[n,:] @ kv) / z[bh][n]   (WMMA, K=64, per-head 128-row tiles)
// ---------------------------------------------------------------------------
#define YP 72  // LDS pitch (bf16) for 64-wide K tiles

__global__ __launch_bounds__(256) void y_wmma_k(const __bf16* __restrict__ Qf,
                                                const float* __restrict__ KV,
                                                const float* __restrict__ Z,
                                                __bf16* __restrict__ Y) {
  const int bh = blockIdx.y;
  const int b = bh >> 4, h = bh & 15;
  const int m0 = blockIdx.x * 128;
  const int tid = threadIdx.x, lane = tid & 31, w = tid >> 5;
  __shared__ __bf16 lQ[128 * YP];   // [m][c]
  __shared__ __bf16 lKV[64 * YP];   // [d][c]  (transposed kv, bf16)

  {  // stage kv^T
    int c = tid >> 2;
    int d16 = (tid & 3) * 16;
    const float* kvp = KV + (size_t)bh * (Cc * Cc) + c * Cc + d16;
    for (int i = 0; i < 16; ++i) lKV[(d16 + i) * YP + c] = (__bf16)kvp[i];
  }
  for (int it = 0; it < 4; ++it) {  // stage 128x64 q tile
    int r = it * 32 + (tid >> 3);
    int cc = (tid & 7) * 8;
    u32x4 v = *(const u32x4*)(Qf + ((size_t)(b * Nseq + m0 + r)) * Dm + h * Cc + cc);
    *(u32x4*)(lQ + r * YP + cc) = v;
  }
  __syncthreads();

  const int r0 = w * 16;
  f32x8 acc[4];
  for (int j = 0; j < 4; ++j) acc[j] = zero8();
  for (int kk = 0; kk < 64; kk += 32) {
    bf16x16 a  = frag_ld_a(lQ, r0, YP, kk, lane);
    bf16x16 b0 = frag_ld_b(lKV, 0,  YP, kk, lane);
    bf16x16 b1 = frag_ld_b(lKV, 16, YP, kk, lane);
    bf16x16 b2 = frag_ld_b(lKV, 32, YP, kk, lane);
    bf16x16 b3 = frag_ld_b(lKV, 48, YP, kk, lane);
    acc[0] = wmma_bf16(a, b0, acc[0]);
    acc[1] = wmma_bf16(a, b1, acc[1]);
    acc[2] = wmma_bf16(a, b2, acc[2]);
    acc[3] = wmma_bf16(a, b3, acc[3]);
  }

  const int mhalf = (lane >> 4) * 8, col = lane & 15;
  for (int j = 0; j < 4; ++j)
    for (int e = 0; e < 8; ++e) {
      int m = r0 + mhalf + e;
      float z = Z[(size_t)bh * Nseq + m0 + m];
      Y[((size_t)(b * Nseq + m0 + m)) * Dm + h * Cc + j * 16 + col] = (__bf16)(acc[j][e] / z);
    }
}

// ---------------------------------------------------------------------------
// Host orchestration
// ---------------------------------------------------------------------------
extern "C" void kernel_launch(void* const* d_in, const int* in_sizes, int n_in,
                              void* d_out, int out_size, void* d_ws, size_t ws_size,
                              hipStream_t stream) {
  (void)in_sizes; (void)n_in; (void)out_size; (void)ws_size;
  const float* q     = (const float*)d_in[0];
  const float* k     = (const float*)d_in[1];
  const float* v     = (const float*)d_in[2];
  const float* Wq    = (const float*)d_in[3];
  const float* Wk    = (const float*)d_in[4];
  const float* Wv    = (const float*)d_in[5];
  const float* Wproj = (const float*)d_in[6];
  const float* ff    = (const float*)d_in[7];
  float* out = (float*)d_out;

  char* ws = (char*)d_ws;
  const size_t MB = 1024ull * 1024ull;
  __bf16* Qb  = (__bf16*)(ws + 0 * MB);     // 64 MB: q bf16, later qf
  __bf16* Kb  = (__bf16*)(ws + 64 * MB);    // 64 MB: k bf16, later kf
  __bf16* Vb  = (__bf16*)(ws + 128 * MB);   // 64 MB: v bf16, later vp bf16
  __bf16* Wqb = (__bf16*)(ws + 192 * MB);
  __bf16* Wkb = (__bf16*)(ws + 194 * MB);
  __bf16* Wvb = (__bf16*)(ws + 196 * MB);
  __bf16* Wpb = (__bf16*)(ws + 198 * MB);
  float*  P0  = (float*)(ws + 200 * MB);    // 128 MB f32 scratch (qp/kp/vp)
  __bf16* Yb  = (__bf16*)(ws + 200 * MB);   // alias: y bf16 after P0 is dead
  float*  KVb = (float*)(ws + 328 * MB);    // 1 MB
  float*  KMb = (float*)(ws + 329 * MB);    // 16 KB
  float*  Zb  = (float*)(ws + 330 * MB);    // 2 MB

  const long NQ = (long)Bsz * Nseq * Dm;    // 33,554,432
  const long NW = (long)Dm * Dm;            // 1,048,576
  const int  Mrows = Bsz * Nseq;            // 32768

  // 1) convert inputs/weights to bf16
  cvt_f32_bf16_k<<<NQ / 1024, 256, 0, stream>>>(q, Qb, NQ);
  cvt_f32_bf16_k<<<NQ / 1024, 256, 0, stream>>>(k, Kb, NQ);
  cvt_f32_bf16_k<<<NQ / 1024, 256, 0, stream>>>(v, Vb, NQ);
  cvt_f32_bf16_k<<<NW / 1024, 256, 0, stream>>>(Wq, Wqb, NW);
  cvt_f32_bf16_k<<<NW / 1024, 256, 0, stream>>>(Wk, Wkb, NW);
  cvt_f32_bf16_k<<<NW / 1024, 256, 0, stream>>>(Wv, Wvb, NW);
  cvt_f32_bf16_k<<<NW / 1024, 256, 0, stream>>>(Wproj, Wpb, NW);

  dim3 ggrid(Dm / GBN, Mrows / GBM);  // (8, 256)

  // 2) qp = q @ Wq^T ; feature map -> qf (into Qb)
  gemm_bf16_wmma_k<<<ggrid, 256, 0, stream>>>(Qb, Wqb, P0, Mrows, Dm, Dm);
  feature_map_k<<<Mrows, 256, 0, stream>>>(P0, Qb, ff);

  // 3) kp = k @ Wk^T ; feature map -> kf (into Kb)
  gemm_bf16_wmma_k<<<ggrid, 256, 0, stream>>>(Kb, Wkb, P0, Mrows, Dm, Dm);
  feature_map_k<<<Mrows, 256, 0, stream>>>(P0, Kb, ff);

  // 4) vp = v @ Wv^T ; convert to bf16 (into Vb)
  gemm_bf16_wmma_k<<<ggrid, 256, 0, stream>>>(Vb, Wvb, P0, Mrows, Dm, Dm);
  cvt_f32_bf16_k<<<NQ / 1024, 256, 0, stream>>>(P0, Vb, NQ);

  // 5) per-head statistics + linear attention
  kmean_k<<<Bsz * Hh, 256, 0, stream>>>(Kb, KMb);
  zinv_k<<<dim3(Nseq / 256, Bsz * Hh), 256, 0, stream>>>(Qb, KMb, Zb);
  kv_wmma_k<<<Bsz * Hh, 256, 0, stream>>>(Kb, Vb, KVb);
  y_wmma_k<<<dim3(Nseq / 128, Bsz * Hh), 256, 0, stream>>>(Qb, KVb, Zb, Yb);

  // 6) out = y @ Wproj^T  (f32 output)
  gemm_bf16_wmma_k<<<ggrid, 256, 0, stream>>>(Yb, Wpb, out, Mrows, Dm, Dm);
}
